// InferTree_80023830659210
// MI455X (gfx1250) — matrix operand order repaired and checked by reading, not compiled, not run on hardware
//
#include <hip/hip_runtime.h>
#include <hip/hip_bf16.h>

// ---------------------------------------------------------------------------
// Soft decision tree inference on gfx1250 (MI455X).
//   logits[b,n,k] = x[b,:] @ W[n,:,k] + (node_weight[n,:] @ W[n,:,k] + b[n,k])
//   prob = softmax_k(logits); path-prob product down 3 levels -> out[8192,1000]
// Main GEMM runs on v_wmma_f32_16x16x32_bf16 (K+1=11 padded to 16 columns).
// ---------------------------------------------------------------------------

typedef __attribute__((ext_vector_type(16))) __bf16 v16bf;
typedef __attribute__((ext_vector_type(8)))  __bf16 v8bf;
typedef __attribute__((ext_vector_type(8)))  float  v8f;

#define NN      111      // nodes
#define DIM     2048
#define KB      64       // DIM / 32
#define BATCH   8192
#define PSTRIDE 12       // LDS prob stride (cols 1..10 used)

__device__ __forceinline__ unsigned short f2bf(float f) {
    unsigned int u = __float_as_uint(f);
    unsigned int r = (u + 0x7FFFu + ((u >> 16) & 1u)) >> 16;   // RNE
    return (unsigned short)r;
}

// ---- prep 1: x f32 -> bf16, row-major [8192][2048] --------------------------
__global__ __launch_bounds__(256) void k_xb(const float* __restrict__ x,
                                            unsigned short* __restrict__ XB) {
    int g = blockIdx.x * 256 + threadIdx.x;        // 2,097,152 threads, 8 elems each
    const float4* xv = (const float4*)x;
    float4 a = xv[2 * g];
    float4 b = xv[2 * g + 1];
    unsigned int p0 = (unsigned int)f2bf(a.x) | ((unsigned int)f2bf(a.y) << 16);
    unsigned int p1 = (unsigned int)f2bf(a.z) | ((unsigned int)f2bf(a.w) << 16);
    unsigned int p2 = (unsigned int)f2bf(b.x) | ((unsigned int)f2bf(b.y) << 16);
    unsigned int p3 = (unsigned int)f2bf(b.z) | ((unsigned int)f2bf(b.w) << 16);
    ((uint4*)XB)[g] = make_uint4(p0, p1, p2, p3);
}

// ---- prep 2: W [111][2048][11] f32 -> per-lane WMMA B-fragments -------------
// WB[n][kb][lane][j]: col = lane&15 (0 for col>=11), d = kb*32 + (lane>=16?16:0) + j
__global__ __launch_bounds__(256) void k_wb(const float* __restrict__ W,
                                            unsigned short* __restrict__ WB) {
    int g = blockIdx.x * 256 + threadIdx.x;        // 111*2048*16 = 3,637,248 exact
    int j    = g & 15;
    int lane = (g >> 4) & 31;
    int kb   = (g >> 9) & 63;
    int n    = g >> 15;
    int d    = kb * 32 + ((lane >> 4) << 4) + j;
    int col  = lane & 15;
    float v  = (col < 11) ? W[((size_t)n * DIM + d) * 11 + col] : 0.0f;
    WB[g] = f2bf(v);
}

// ---- prep 3: per-node constant bias: node_weight . W + b  (f32, [111][16]) --
__global__ __launch_bounds__(256) void k_bias(const float* __restrict__ nodew,
                                              const float* __restrict__ W,
                                              const float* __restrict__ b,
                                              float* __restrict__ bias) {
    __shared__ float red[256];
    int n = blockIdx.x, t = threadIdx.x;
    int col = t & 15, dt = t >> 4;                 // 16 threads per column
    float acc = 0.0f;
    if (col < 11) {
        for (int d = dt; d < DIM; d += 16)
            acc += nodew[(size_t)n * DIM + d] * W[((size_t)n * DIM + d) * 11 + col];
    }
    red[t] = acc;
    __syncthreads();
    for (int s = 128; s >= 16; s >>= 1) {
        if (t < s) red[t] += red[t + s];
        __syncthreads();
    }
    if (t < 16)
        bias[n * 16 + t] = (t < 11) ? (red[t] + b[n * 11 + t]) : -1.0e30f;
}

// ---- main fused kernel: WMMA GEMM + softmax + tree propagation --------------
// grid = 512 (batch tiles of 16 rows), block = 256 (8 waves, nodes strided by 8)
__global__ __launch_bounds__(256) void k_main(const unsigned short* __restrict__ XB,
                                              const unsigned short* __restrict__ WB,
                                              const float* __restrict__ bias,
                                              float* __restrict__ out) {
    extern __shared__ float probLDS[];             // [111][16 rows][PSTRIDE]
    const int lane  = threadIdx.x & 31;
    const int wave  = threadIdx.x >> 5;
    const int tile  = blockIdx.x;
    const int l16   = lane & 15;                   // A-row / B,C-column for this lane
    const int half  = lane >> 4;

    // A-fragment base: row = l16 of the tile; contiguous K order matching WB:
    // lanes 0-15 pull K {0..15}, lanes 16-31 pull K {16..31}.
    const unsigned short* xrow =
        XB + ((size_t)tile * 16 + l16) * DIM + (half << 4);
    const v16bf* WBv = (const v16bf*)WB;

    for (int n = wave; n < NN; n += 8) {
        v8f acc = {};
        const v16bf* wbn = WBv + (size_t)n * (KB * 32) + lane;
        for (int kb = 0; kb < KB; ++kb) {
            const v8bf* xp = (const v8bf*)(xrow + kb * 32);
            v8bf lo = xp[0];                       // K base .. base+7
            v8bf hi = xp[1];                       // K base+8 .. base+15
            v16bf afrag = __builtin_shufflevector(
                lo, hi, 0, 1, 2, 3, 4, 5, 6, 7, 8, 9, 10, 11, 12, 13, 14, 15);
            v16bf bfrag = wbn[kb * 32];            // contiguous 32B per lane
            acc = __builtin_amdgcn_wmma_f32_16x16x32_bf16(
                false, afrag, false, bfrag, (short)0, acc, false, false);
        }
        // C layout: acc[r] = logits[row = r + 8*half][col = l16]
        float bb = bias[n * 16 + l16];
#pragma unroll
        for (int r = 0; r < 8; ++r) {
            float v = acc[r] + bb;
            float m = v;                           // max over 16 cols (16-lane halves)
            m = fmaxf(m, __shfl_xor(m, 1, 32));
            m = fmaxf(m, __shfl_xor(m, 2, 32));
            m = fmaxf(m, __shfl_xor(m, 4, 32));
            m = fmaxf(m, __shfl_xor(m, 8, 32));
            float e = __expf(v - m);
            float s = e;
            s += __shfl_xor(s, 1, 32);
            s += __shfl_xor(s, 2, 32);
            s += __shfl_xor(s, 4, 32);
            s += __shfl_xor(s, 8, 32);
            float p = e / s;
            int row = r + (half << 3);
            if (l16 >= 1 && l16 <= 10)             // only slots 1..10 feed the tree
                probLDS[(n * 16 + row) * PSTRIDE + l16] = p;
        }
    }
    __syncthreads();

    // Path-probability propagation: 16 rows x 100 level-2 nodes = 1600 tasks.
    for (int task = threadIdx.x; task < 1600; task += 256) {
        int row = task / 100;
        int c   = task % 100;                      // level-2 node = 11 + c
        int i   = c / 10, j2 = c % 10;             // parent chain: root -> (1+i) -> (11+c)
        float pp = probLDS[(0 * 16 + row) * PSTRIDE + (1 + i)] *
                   probLDS[((1 + i) * 16 + row) * PSTRIDE + (1 + j2)];
        int n2 = 11 + c;
        float* orow = out + ((size_t)tile * 16 + row) * 1000 + c * 10;
#pragma unroll
        for (int j = 0; j < 10; ++j)
            orow[j] = probLDS[(n2 * 16 + row) * PSTRIDE + (1 + j)] * pp;
    }
}

extern "C" void kernel_launch(void* const* d_in, const int* in_sizes, int n_in,
                              void* d_out, int out_size, void* d_ws, size_t ws_size,
                              hipStream_t stream) {
    const float* x     = (const float*)d_in[0];
    // d_in[1] = labels (unused in forward)
    const float* nodew = (const float*)d_in[2];
    const float* W     = (const float*)d_in[3];
    const float* b     = (const float*)d_in[4];
    float* out = (float*)d_out;

    // workspace layout
    unsigned short* XB = (unsigned short*)d_ws;                 // 8192*2048 bf16 = 32 MiB
    unsigned short* WB = XB + (size_t)BATCH * DIM;              // 111*2048*16 bf16 ~ 7 MiB
    float* bias = (float*)(WB + (size_t)NN * DIM * 16);         // 111*16 f32

    k_xb  <<<(BATCH * DIM / 8) / 256, 256, 0, stream>>>(x, XB);
    k_wb  <<<(NN * DIM * 16) / 256,   256, 0, stream>>>(W, WB);
    k_bias<<<NN,                      256, 0, stream>>>(nodew, W, b, bias);

    size_t shmem = (size_t)NN * 16 * PSTRIDE * sizeof(float);   // 85,248 B
    k_main<<<BATCH / 16, 256, shmem, stream>>>(XB, WB, bias, out);
}
